// LinearQuant_56272661512689
// MI455X (gfx1250) — 8-bit matrix operand order repaired and checked by reading, compiled, not run on hardware
//
#include <hip/hip_runtime.h>
#include <stdint.h>


typedef __attribute__((ext_vector_type(8))) int v8i;
typedef int v4i_vs __attribute__((vector_size(16)));
typedef __attribute__((address_space(1))) v4i_vs gv4i_t;
typedef __attribute__((address_space(3))) v4i_vs lv4i_t;

#define BM 128
#define BN 128
#define KB 64
#define LDB 80
#define TILE_BYTES (BM * LDB)
#define BUF_STRIDE (2 * TILE_BYTES)

#if __has_builtin(__builtin_amdgcn_global_load_async_to_lds_b128)
#define ASYNC_LDS 1
#else
#define ASYNC_LDS 0
#endif

__device__ __forceinline__ void stage16(const uint8_t* gsrc, uint8_t* ldst) {
#if ASYNC_LDS
    __builtin_amdgcn_global_load_async_to_lds_b128((gv4i_t*)gsrc, (lv4i_t*)ldst,
                                                    0,  0);
#else
    *(int4*)ldst = *(const int4*)gsrc;
#endif
}

__device__ __forceinline__ void stage_drain() {
#if ASYNC_LDS
#if __has_builtin(__builtin_amdgcn_s_wait_asynccnt)
    __builtin_amdgcn_s_wait_asynccnt(0);
#else
    asm volatile("s_wait_asynccnt 0x0" ::: "memory");
#endif
#endif
}

__global__ __launch_bounds__(256) void lq_pack_kernel(const int4* __restrict__ src,
                                                      unsigned* __restrict__ dst,
                                                      int n4) {
    int i = blockIdx.x * 256 + threadIdx.x;
    if (i < n4) {
        int4 v = src[i];
        unsigned p = (unsigned)(v.x & 0xFF)
                   | ((unsigned)(v.y & 0xFF) << 8)
                   | ((unsigned)(v.z & 0xFF) << 16)
                   | ((unsigned)(v.w & 0xFF) << 24);
        dst[i] = p;
    }
}

__global__ __launch_bounds__(256) void lq_gemm_kernel(const uint8_t* __restrict__ Apk,
                                                      const uint8_t* __restrict__ Wpk,
                                                      const int* __restrict__ bias,
                                                      const float* __restrict__ scale,
                                                      float* __restrict__ out,
                                                      int Mtot, int Ntot, int Ktot) {
    __shared__ __align__(16) uint8_t smem[4 * TILE_BYTES];

    const int tid  = threadIdx.x;
    const int lane = tid & 31;
    const int wave = tid >> 5;
    const int wm   = wave & 3;
    const int wn   = wave >> 2;

    const int blockM = blockIdx.y * BM;
    const int blockN = blockIdx.x * BN;

    const int stRow  = tid >> 2;
    const int stKoff = (tid & 3) * 16;

    const uint8_t* aGlob = Apk + (size_t)blockM * Ktot;
    const uint8_t* bGlob = Wpk + (size_t)blockN * Ktot;

    const int frow = lane & 15;
    const int fhi  = lane >> 4;

    v8i acc[2][4];
#pragma unroll
    for (int mt = 0; mt < 2; ++mt)
#pragma unroll
        for (int nt = 0; nt < 4; ++nt)
            acc[mt][nt] = {};

    const int kSteps = Ktot / KB;

    {
#pragma unroll
        for (int s = 0; s < 2; ++s) {
            int r = stRow + s * 64;
            stage16(aGlob + (size_t)r * Ktot + stKoff,
                    smem + r * LDB + stKoff);
            stage16(bGlob + (size_t)r * Ktot + stKoff,
                    smem + TILE_BYTES + r * LDB + stKoff);
        }
    }
    stage_drain();
    __syncthreads();

    for (int kk = 0; kk < kSteps; ++kk) {
        const int cur = kk & 1;
        uint8_t* Acur = smem + cur * BUF_STRIDE;
        uint8_t* Bcur = Acur + TILE_BYTES;

        if (kk + 1 < kSteps) {
            uint8_t* Anxt = smem + (cur ^ 1) * BUF_STRIDE;
            uint8_t* Bnxt = Anxt + TILE_BYTES;
#pragma unroll
            for (int s = 0; s < 2; ++s) {
                int r = stRow + s * 64;
                stage16(aGlob + (size_t)r * Ktot + (kk + 1) * KB + stKoff,
                        Anxt + r * LDB + stKoff);
                stage16(bGlob + (size_t)r * Ktot + (kk + 1) * KB + stKoff,
                        Bnxt + r * LDB + stKoff);
            }
        }

        v8i afrag[2];
#pragma unroll
        for (int mt = 0; mt < 2; ++mt) {
            const uint8_t* base = Acur + (wm * 32 + mt * 16 + frow) * LDB + 16 * fhi;
            int4* ap = (int4*)&afrag[mt];
            ap[0] = *(const int4*)(base);
            ap[1] = *(const int4*)(base + 32);
        }

        v8i bfrag[4];
#pragma unroll
        for (int nt = 0; nt < 4; ++nt) {
            const uint8_t* base = Bcur + (wn * 64 + nt * 16 + frow) * LDB + 16 * fhi;
            int4* bp = (int4*)&bfrag[nt];
            bp[0] = *(const int4*)(base);
            bp[1] = *(const int4*)(base + 32);
        }

#pragma unroll
        for (int mt = 0; mt < 2; ++mt)
#pragma unroll
            for (int nt = 0; nt < 4; ++nt)
                acc[mt][nt] = __builtin_amdgcn_wmma_i32_16x16x64_iu8(
                     true, afrag[mt],
                     true, bfrag[nt],
                    acc[mt][nt],
                     false,  false);

        stage_drain();
        __syncthreads();
    }

#pragma unroll
    for (int nt = 0; nt < 4; ++nt) {
        const int col = blockN + wn * 64 + nt * 16 + frow;
        const int   bv = bias[col];
        const float sv = scale[col];
#pragma unroll
        for (int mt = 0; mt < 2; ++mt) {
            const int rbase = blockM + wm * 32 + mt * 16 + 8 * fhi;
#pragma unroll
            for (int r = 0; r < 8; ++r) {
                out[(size_t)(rbase + r) * Ntot + col] =
                    (float)(acc[mt][nt][r] + bv) * sv;
            }
        }
    }
}

extern "C" void kernel_launch(void* const* d_in, const int* in_sizes, int n_in,
                              void* d_out, int out_size, void* d_ws, size_t ws_size,
                              hipStream_t stream) {
    const int*   x     = (const int*)d_in[0];
    const int*   w     = (const int*)d_in[1];
    const float* scale = (const float*)d_in[2];
    const int*   bias  = (const int*)d_in[3];
    float*       out   = (float*)d_out;

    const int N = in_sizes[2];
    const int K = in_sizes[1] / N;
    const int M = in_sizes[0] / K;

    uint8_t* Apk = (uint8_t*)d_ws;
    uint8_t* Wpk = Apk + (size_t)M * K;

    const int an4 = (M * K) / 4;
    const int wn4 = (N * K) / 4;
    lq_pack_kernel<<<(an4 + 255) / 256, 256, 0, stream>>>((const int4*)x, (unsigned*)Apk, an4);
    lq_pack_kernel<<<(wn4 + 255) / 256, 256, 0, stream>>>((const int4*)w, (unsigned*)Wpk, wn4);

    dim3 grid(N / BN, M / BM);
    lq_gemm_kernel<<<grid, 256, 0, stream>>>(Apk, Wpk, bias, scale, out, M, N, K);
}
